// DynamicNetwork_61813169324318
// MI455X (gfx1250) — matrix operand order repaired and checked by reading, not compiled, not run on hardware
//
#include <hip/hip_runtime.h>
#include <hip/hip_bf16.h>

typedef __attribute__((ext_vector_type(16))) __bf16 v16bf;
typedef __attribute__((ext_vector_type(8)))  float  v8f;

#define HN 1024      // nodes per level (K dimension and hidden N)
#define BN 1024      // batch (M dimension)
#define FANIN 64
#define NOUTN 256

// ---------------------------------------------------------------------------
// Convert f32 -> bf16 (streaming)
// ---------------------------------------------------------------------------
__global__ __launch_bounds__(256) void f32_to_bf16_kernel(
    const float* __restrict__ in, __hip_bfloat16* __restrict__ out, int n) {
  int i = blockIdx.x * blockDim.x + threadIdx.x;
  if (i < n) out[i] = __float2bfloat16(in[i]);
}

// ---------------------------------------------------------------------------
// Densify one level's edge list into a row-major bf16 matrix Wd[row][HN].
// One wave per dst row: zero a 4KB LDS f32 row, ds_add_f32-scatter the 64
// edge weights (handles duplicate src collisions), then stream out as bf16.
// ---------------------------------------------------------------------------
__global__ __launch_bounds__(256) void densify_kernel(
    const int*   __restrict__ src,   // [rows*FANIN]
    const float* __restrict__ w,     // [rows*FANIN]
    __hip_bfloat16* __restrict__ Wd) // [rows][HN]
{
  __shared__ float rowbuf[8][HN];            // 8 waves/block, 4KB each = 32KB
  const int wv   = threadIdx.x >> 5;
  const int lane = threadIdx.x & 31;
  const int row  = blockIdx.x * 8 + wv;
  float* rb = rowbuf[wv];

  for (int j = lane; j < HN; j += 32) rb[j] = 0.0f;
  __syncthreads();

  const int e0 = row * FANIN;
  for (int k = lane; k < FANIN; k += 32) {
    const int e = e0 + k;
    atomicAdd(&rb[src[e]], w[e]);            // ds_add_f32
  }
  __syncthreads();

  __hip_bfloat16* out = Wd + (size_t)row * HN;
  for (int j = lane; j < HN; j += 32) out[j] = __float2bfloat16(rb[j]);
}

// ---------------------------------------------------------------------------
// Dense bf16 GEMM:  out[M=BN, N=NOUT] = act( V[BN,HN] @ Wd[NOUT,HN]^T )
// One wave -> 32(M) x 64(N) tile: 8 accumulators of v_wmma_f32_16x16x32_bf16,
// 2 A fragments per k-step, each B fragment reused by 2 WMMAs
// (1.5 b128 loads per WMMA vs 2.5 for a 16x64 tile).
// A fragment: lane l<16 holds M=l, K in {0..7}u{16..23}; l>=16: K {8..15}u{24..31}
// B fragment: lane l<16 holds N=l, K in {0..7}u{16..23}; l>=16: K {8..15}u{24..31}
// D: VGPR v -> M = v + 8*(lane>=16), N = lane&15
// ---------------------------------------------------------------------------
template <int NOUT, bool RELU>
__global__ __launch_bounds__(256) void gemm_wmma_kernel(
    const __hip_bfloat16* __restrict__ V,   // [BN][HN] bf16 row-major
    const __hip_bfloat16* __restrict__ Wd,  // [NOUT][HN] bf16 row-major
    void* __restrict__ outp)
{
  union AU { v16bf v; uint4 q[2]; };

  const int lane  = threadIdx.x & 31;
  const int l15   = lane & 15;
  const int lhalf = lane >> 4;                         // 0 or 1
  const int gw    = blockIdx.x * (blockDim.x >> 5) + (threadIdx.x >> 5);
  const int ngrp  = NOUT / 64;
  const int m0    = (gw / ngrp) * 32;
  const int n0    = (gw % ngrp) * 64;

  v8f acc[2][4];
#pragma unroll
  for (int r = 0; r < 2; ++r)
#pragma unroll
    for (int t = 0; t < 4; ++t) acc[r][t] = (v8f){};

  const uint4* arow0 = (const uint4*)(V + (size_t)(m0 +  0 + l15) * HN);
  const uint4* arow1 = (const uint4*)(V + (size_t)(m0 + 16 + l15) * HN);
  const uint4* brow[4];
#pragma unroll
  for (int t = 0; t < 4; ++t)
    brow[t] = (const uint4*)(Wd + (size_t)(n0 + 16 * t + l15) * HN);

  for (int k0 = 0; k0 < HN; k0 += 32) {
    const int q  = k0 >> 3;            // uint4 index of k0 within a row
    const int bq = q + lhalf;

    AU a0, a1;
    a0.q[0] = arow0[q + lhalf];        // K = k0 + lhalf*8 .. +7
    a0.q[1] = arow0[q + 2 + lhalf];    // K = k0 + 16 + lhalf*8 .. +7
    a1.q[0] = arow1[q + lhalf];
    a1.q[1] = arow1[q + 2 + lhalf];

#pragma unroll
    for (int t = 0; t < 4; ++t) {
      AU bb;
      bb.q[0] = brow[t][bq];
      bb.q[1] = brow[t][bq + 2];
      acc[0][t] = __builtin_amdgcn_wmma_f32_16x16x32_bf16(
          false, a0.v, false, bb.v, (short)0, acc[0][t], false, false);
      acc[1][t] = __builtin_amdgcn_wmma_f32_16x16x32_bf16(
          false, a1.v, false, bb.v, (short)0, acc[1][t], false, false);
    }
  }

  if (RELU) {
    __hip_bfloat16* out = (__hip_bfloat16*)outp;
#pragma unroll
    for (int r = 0; r < 2; ++r)
#pragma unroll
      for (int t = 0; t < 4; ++t)
#pragma unroll
        for (int v = 0; v < 8; ++v) {
          const size_t mm = (size_t)(m0 + 16 * r + v + lhalf * 8) * NOUT;
          out[mm + n0 + 16 * t + l15] = __float2bfloat16(fmaxf(acc[r][t][v], 0.0f));
        }
  } else {
    float* out = (float*)outp;
#pragma unroll
    for (int r = 0; r < 2; ++r)
#pragma unroll
      for (int t = 0; t < 4; ++t)
#pragma unroll
        for (int v = 0; v < 8; ++v) {
          const size_t mm = (size_t)(m0 + 16 * r + v + lhalf * 8) * NOUT;
          out[mm + n0 + 16 * t + l15] = acc[r][t][v];
        }
  }
}

// ---------------------------------------------------------------------------
// Launch: convert x -> bf16, then per level: densify W, WMMA GEMM (+ReLU),
// ping-pong vals; finally densify output W and do linear f32-out GEMM.
// Inputs (setup_inputs order): x, src_h, dst_h, w_h, src_out, dst_out, w_out
// dst_h / dst_out are the canonical tiling (dst = edge/64) -> not needed.
// ---------------------------------------------------------------------------
extern "C" void kernel_launch(void* const* d_in, const int* in_sizes, int n_in,
                              void* d_out, int out_size, void* d_ws, size_t ws_size,
                              hipStream_t stream) {
  const float* x       = (const float*)d_in[0];
  const int*   src_h   = (const int*)  d_in[1];
  const float* w_h     = (const float*)d_in[3];
  const int*   src_out = (const int*)  d_in[4];
  const float* w_out   = (const float*)d_in[6];

  const int EH = HN * FANIN;  // 65536 edges per hidden level

  __hip_bfloat16* valsA = (__hip_bfloat16*)d_ws;          // 2 MB
  __hip_bfloat16* valsB = valsA + (size_t)BN * HN;        // 2 MB
  __hip_bfloat16* Wd    = valsB + (size_t)BN * HN;        // 2 MB (reused)

  f32_to_bf16_kernel<<<(BN * HN) / 256, 256, 0, stream>>>(x, valsA, BN * HN);

  __hip_bfloat16* cur = valsA;
  __hip_bfloat16* nxt = valsB;
  for (int l = 0; l < 4; ++l) {
    densify_kernel<<<HN / 8, 256, 0, stream>>>(src_h + (size_t)l * EH,
                                               w_h   + (size_t)l * EH, Wd);
    // (1024/32) M-tiles x (1024/64) N-groups = 512 waves = 64 blocks of 8 waves
    gemm_wmma_kernel<HN, true><<<64, 256, 0, stream>>>(cur, Wd, nxt);
    __hip_bfloat16* t = cur; cur = nxt; nxt = t;
  }

  densify_kernel<<<NOUTN / 8, 256, 0, stream>>>(src_out, w_out, Wd);
  // (1024/32) M-tiles x (256/64) N-groups = 128 waves = 16 blocks
  gemm_wmma_kernel<NOUTN, false><<<16, 256, 0, stream>>>(cur, Wd, d_out);
}
